// VisualDialog_76940044140918
// MI455X (gfx1250) — matrix operand order repaired and checked by reading, not compiled, not run on hardware
//
#include <hip/hip_runtime.h>

// ---------------------------------------------------------------------------
// VisualDialog on MI455X (gfx1250, wave32, WMMA bf16 16x16x32, f32 accum)
//
// decoder "history" only feeds the output projection -> N=512-question decode
// is batch-parallel; recurrence is per batch row (block-local, no grid sync).
// Dominant GEMM: logits[32768,1000] = feat[32768,896] @ Wout^T (~59 GFLOP).
// Round 2: 32-row output block tile (2x B-fragment reuse, halves L2 traffic on
// the dominant GEMM) + GLOBAL_LOAD_ASYNC_TO_LDS_B128 staging (ASYNCcnt path).
// ---------------------------------------------------------------------------

typedef __attribute__((ext_vector_type(8)))  __bf16 v8bf;
typedef __attribute__((ext_vector_type(16))) __bf16 v16bf;
typedef __attribute__((ext_vector_type(8)))  float  v8f;

#define HID   128
#define EMB   100
#define VOC   1000
#define FDIM  512
#define NB    512
#define TQL   64
#define TAL   64
#define GATES 512          // 4*HID
#define KOUT  896          // 3*HID + FDIM
#define KT_OUT 28          // 896/32
#define NT_OUT 63          // 1008/16 (VOC padded to 1008)

// ------------------------- CDNA5 async LDS helpers -------------------------

typedef __attribute__((address_space(3))) void lds_void;

static __device__ inline void async_ld_b128(void* lds_ptr, const void* gptr) {
  unsigned l = (unsigned)(unsigned long long)(lds_void*)lds_ptr;
  asm volatile("global_load_async_to_lds_b128 %0, %1, off"
               :: "v"(l), "v"(gptr) : "memory");
}
static __device__ inline void wait_async0() {
  asm volatile("s_wait_asynccnt 0" ::: "memory");
}

// ------------------------------- WMMA helpers ------------------------------

static __device__ inline v8f wmma_bf16(v16bf a, v16bf b, v8f c) {
  return __builtin_amdgcn_wmma_f32_16x16x32_bf16(false, a, false, b,
                                                 (short)0, c, false, false);
}

// A fragment (16x32 bf16) from a row-major [16, ld] buffer (LDS or global).
static __device__ inline v16bf load_a_frag(const __bf16* base, int ld, int kt, int lane) {
  const int row  = lane & 15;
  const int half = lane >> 4;
  const __bf16* p = base + row * ld + kt * 32 + half * 8;
  v8bf lo = *(const v8bf*)p;
  v8bf hi = *(const v8bf*)(p + 16);
  v16bf a;
#pragma unroll
  for (int j = 0; j < 8; ++j) { a[j] = lo[j]; a[j + 8] = hi[j]; }
  return a;
}

// B fragment (32x16 bf16) from pre-swizzled weights: contiguous 32B per lane.
static __device__ inline v16bf load_b_frag(const __bf16* wsw, int kt, int nt, int NT, int lane) {
  const __bf16* p = wsw + (((size_t)(kt * NT + nt) * 32 + lane) << 4);
  return *(const v16bf*)p;
}

static __device__ inline float sigm(float x) { return 1.f / (1.f + __expf(-x)); }
static __device__ inline float tanh_fast(float x) {
  float ax = fabsf(x);
  float e  = __expf(-2.f * ax);
  float t  = (1.f - e) / (1.f + e);
  return copysignf(t, x);
}

// ------------------------------- prep kernels ------------------------------

// W [outdim, indim] f32 -> fragment-swizzled bf16 B layout [KT][NT][32][16]
__global__ __launch_bounds__(256) void k_swizzleW(const float* __restrict__ W,
                                                  __bf16* __restrict__ dst,
                                                  int outdim, int indim, int KT, int NT) {
  int idx = blockIdx.x * blockDim.x + threadIdx.x;
  int total = KT * NT * 512;
  if (idx >= total) return;
  int j    = idx & 15;
  int lane = (idx >> 4) & 31;
  int rest = idx >> 9;
  int nt   = rest % NT;
  int kt   = rest / NT;
  int k = kt * 32 + (j >> 3) * 16 + (lane >> 4) * 8 + (j & 7);     // K index
  int n = nt * 16 + (lane & 15);              // N index
  float v = (k < indim && n < outdim) ? W[(size_t)n * indim + k] : 0.f;
  dst[idx] = (__bf16)v;
}

// src [rows, cols] f32 -> dst [rows, ld] bf16 (zero-padded columns)
__global__ __launch_bounds__(256) void k_pad(const float* __restrict__ src,
                                             __bf16* __restrict__ dst,
                                             int rows, int cols, int ld) {
  int idx = blockIdx.x * blockDim.x + threadIdx.x;
  if (idx >= rows * ld) return;
  int r = idx / ld, c = idx % ld;
  dst[idx] = (c < cols) ? (__bf16)src[(size_t)r * cols + c] : (__bf16)0.f;
}

// ------------------------------- LSTM layer --------------------------------
// One 16-row tile, 8 waves. Wave w owns gate N-tiles 4w..4w+3.
static __device__ void lstm_layer(const __bf16* xA, __bf16* hS, float* cS,
                                  const __bf16* WihSw, const __bf16* WhhSw,
                                  const float* bias, float* zb,
                                  int wave, int lane) {
  v8f acc[4];
#pragma unroll
  for (int q = 0; q < 4; ++q) {
    float bv = bias[(wave * 4 + q) * 16 + (lane & 15)];
#pragma unroll
    for (int r = 0; r < 8; ++r) acc[q][r] = bv;
  }
#pragma unroll
  for (int kt = 0; kt < 4; ++kt) {
    v16bf aX = load_a_frag(xA, HID, kt, lane);
    v16bf aH = load_a_frag(hS, HID, kt, lane);
#pragma unroll
    for (int q = 0; q < 4; ++q) {
      int nt = wave * 4 + q;
      acc[q] = wmma_bf16(aX, load_b_frag(WihSw, kt, nt, 32, lane), acc[q]);
      acc[q] = wmma_bf16(aH, load_b_frag(WhhSw, kt, nt, 32, lane), acc[q]);
    }
  }
#pragma unroll
  for (int q = 0; q < 4; ++q) {
    int col = (wave * 4 + q) * 16 + (lane & 15);
#pragma unroll
    for (int r = 0; r < 8; ++r) {
      int row = r + ((lane >> 4) << 3);
      zb[row * GATES + col] = acc[q][r];
    }
  }
  __syncthreads();
  int row = lane & 15;
  int cb  = wave * 16 + ((lane >> 4) << 3);
#pragma unroll
  for (int j = 0; j < 8; ++j) {
    int hc = cb + j;
    float iv = zb[row * GATES + hc];
    float fv = zb[row * GATES + 128 + hc];
    float gv = zb[row * GATES + 256 + hc];
    float ov = zb[row * GATES + 384 + hc];
    float c = cS[row * HID + hc];
    c = sigm(fv) * c + sigm(iv) * tanh_fast(gv);
    float h = sigm(ov) * tanh_fast(c);
    cS[row * HID + hc] = c;
    hS[row * HID + hc] = (__bf16)h;
  }
  __syncthreads();
}

// ------------------------------- encoder -----------------------------------
__global__ __launch_bounds__(256) void k_encoder(
    const int* __restrict__ questions, const __bf16* __restrict__ embB,
    const __bf16* Wih0, const __bf16* Whh0, const float* b0,
    const __bf16* Wih1, const __bf16* Whh1, const float* b1,
    __bf16* __restrict__ outs,
    __bf16* __restrict__ ehb0, float* __restrict__ ec0,
    __bf16* __restrict__ ehb1, float* __restrict__ ec1) {
  __shared__ __bf16 xA[16 * HID];
  __shared__ __bf16 h0s[16 * HID];
  __shared__ __bf16 h1s[16 * HID];
  __shared__ float  zb[16 * GATES];
  __shared__ float  c0s[16 * HID];
  __shared__ float  c1s[16 * HID];
  const int n0 = blockIdx.x * 16;
  const int tid = threadIdx.x, wave = tid >> 5, lane = tid & 31;

  for (int i = tid; i < 16 * HID; i += 256) {
    h0s[i] = (__bf16)0.f; h1s[i] = (__bf16)0.f; c0s[i] = 0.f; c1s[i] = 0.f;
  }
  __syncthreads();

  for (int t = 0; t < TQL; ++t) {
    {  // async embed gather: 256 threads = 16 rows x 16 chunks of 8 bf16
      int row = tid >> 4, c = tid & 15;
      int tok = questions[(n0 + row) * TQL + t];
      async_ld_b128(&xA[row * HID + c * 8], embB + (size_t)tok * HID + c * 8);
      wait_async0();
    }
    __syncthreads();
    lstm_layer(xA, h0s, c0s, Wih0, Whh0, b0, zb, wave, lane);
    lstm_layer(h0s, h1s, c1s, Wih1, Whh1, b1, zb, wave, lane);
    for (int i = tid; i < 16 * HID; i += 256) {
      int row = i >> 7, k = i & 127;
      outs[((size_t)(n0 + row) * TQL + t) * HID + k] = h1s[i];
    }
    __syncthreads();
  }
  for (int i = tid; i < 16 * HID; i += 256) {
    int row = i >> 7, k = i & 127;
    size_t g = (size_t)(n0 + row) * HID + k;
    ehb0[g] = h0s[i]; ehb1[g] = h1s[i]; ec0[g] = c0s[i]; ec1[g] = c1s[i];
  }
}

// ------------------------------- keys GEMM ---------------------------------
__global__ __launch_bounds__(256) void k_keys(const __bf16* __restrict__ outs,
                                              const __bf16* __restrict__ wk,
                                              const float* __restrict__ bk,
                                              __bf16* __restrict__ keys) {
  const int wave = threadIdx.x >> 5, lane = threadIdx.x & 31;
  const int mt = blockIdx.x * 8 + wave;
  const __bf16* arow = outs + (size_t)mt * 16 * HID;
  v8f acc[8];
#pragma unroll
  for (int nt = 0; nt < 8; ++nt) {
    float bv = bk[nt * 16 + (lane & 15)];
#pragma unroll
    for (int r = 0; r < 8; ++r) acc[nt][r] = bv;
  }
#pragma unroll
  for (int kt = 0; kt < 4; ++kt) {
    v16bf a = load_a_frag(arow, HID, kt, lane);
#pragma unroll
    for (int nt = 0; nt < 8; ++nt)
      acc[nt] = wmma_bf16(a, load_b_frag(wk, kt, nt, 8, lane), acc[nt]);
  }
#pragma unroll
  for (int nt = 0; nt < 8; ++nt) {
#pragma unroll
    for (int r = 0; r < 8; ++r) {
      int row = r + ((lane >> 4) << 3);
      int col = nt * 16 + (lane & 15);
      keys[((size_t)mt * 16 + row) * HID + col] = (__bf16)acc[nt][r];
    }
  }
}

// ------------------------------- decoder -----------------------------------
__global__ __launch_bounds__(256) void k_decoder(
    const int* __restrict__ answers, const __bf16* __restrict__ embB,
    const __bf16* Wih0, const __bf16* Whh0, const float* b0,
    const __bf16* Wih1, const __bf16* Whh1, const float* b1,
    const __bf16* __restrict__ eh0i, const float* __restrict__ ec0i,
    const __bf16* __restrict__ eh1i, const float* __restrict__ ec1i,
    const __bf16* __restrict__ keys,
    __bf16* __restrict__ h1ctx,                // [N*TA, 256] = {h1, ctx}
    __bf16* __restrict__ h1last) {             // [N, 128] final h1
  __shared__ __bf16 xA[16 * HID];
  __shared__ __bf16 h0s[16 * HID];
  __shared__ __bf16 h1s[16 * HID];
  __shared__ float  zb[16 * GATES];            // reused as attn probs
  __shared__ float  c0s[16 * HID];
  __shared__ float  c1s[16 * HID];
  const int n0 = blockIdx.x * 16;
  const int tid = threadIdx.x, wave = tid >> 5, lane = tid & 31;
  float* attp = zb;

  for (int i = tid; i < 16 * HID; i += 256) {
    int row = i >> 7, k = i & 127;
    size_t g = (size_t)(n0 + row) * HID + k;
    h0s[i] = eh0i[g]; h1s[i] = eh1i[g]; c0s[i] = ec0i[g]; c1s[i] = ec1i[g];
  }
  __syncthreads();

  for (int t = 0; t < TAL; ++t) {
    {  // async embed gather
      int row = tid >> 4, c = tid & 15;
      int tok = answers[(n0 + row) * TAL + t];
      async_ld_b128(&xA[row * HID + c * 8], embB + (size_t)tok * HID + c * 8);
      wait_async0();
    }
    __syncthreads();
    lstm_layer(xA, h0s, c0s, Wih0, Whh0, b0, zb, wave, lane);
    lstm_layer(h0s, h1s, c1s, Wih1, Whh1, b1, zb, wave, lane);

    // ---- attention: wave w handles rows 2w, 2w+1 ----
#pragma unroll
    for (int rr = 0; rr < 2; ++rr) {
      int row = wave * 2 + rr;
      int n = n0 + row;
      float s0 = 0.f, s1 = 0.f;
      const __bf16* kr0 = keys + ((size_t)(n * TQL + lane)) * HID;
      const __bf16* kr1 = keys + ((size_t)(n * TQL + lane + 32)) * HID;
      for (int kk = 0; kk < 16; ++kk) {
        v8bf k0 = *(const v8bf*)(kr0 + kk * 8);
        v8bf k1 = *(const v8bf*)(kr1 + kk * 8);
#pragma unroll
        for (int j = 0; j < 8; ++j) {
          float hv = (float)h1s[row * HID + kk * 8 + j];
          s0 += (float)k0[j] * hv;
          s1 += (float)k1[j] * hv;
        }
      }
      float m = fmaxf(s0, s1);
#pragma unroll
      for (int off = 16; off > 0; off >>= 1) m = fmaxf(m, __shfl_xor(m, off, 32));
      float e0 = __expf(s0 - m), e1 = __expf(s1 - m);
      float sum = e0 + e1;
#pragma unroll
      for (int off = 16; off > 0; off >>= 1) sum += __shfl_xor(sum, off, 32);
      float inv = 1.f / sum;
      attp[row * 64 + lane] = e0 * inv;
      attp[row * 64 + lane + 32] = e1 * inv;
      float c0v = 0.f, c1v = 0.f, c2v = 0.f, c3v = 0.f;
      for (int tq = 0; tq < TQL; ++tq) {
        float p = attp[row * 64 + tq];
        const __bf16* kp = keys + ((size_t)(n * TQL + tq)) * HID + lane * 4;
        c0v += p * (float)kp[0]; c1v += p * (float)kp[1];
        c2v += p * (float)kp[2]; c3v += p * (float)kp[3];
      }
      __bf16* cd = h1ctx + ((size_t)(n * TAL + t)) * 256 + 128 + lane * 4;
      cd[0] = (__bf16)c0v; cd[1] = (__bf16)c1v;
      cd[2] = (__bf16)c2v; cd[3] = (__bf16)c3v;
    }
    for (int i = tid; i < 16 * HID; i += 256) {
      int row = i >> 7, k = i & 127;
      h1ctx[((size_t)((n0 + row) * TAL + t)) * 256 + k] = h1s[i];
    }
    __syncthreads();
  }
  for (int i = tid; i < 16 * HID; i += 256) {
    int row = i >> 7, k = i & 127;
    h1last[(size_t)(n0 + row) * HID + k] = h1s[i];
  }
}

// --------------------- fused feat-assembly + output GEMM -------------------
// 32-row block tile: each wave keeps two 16-row accumulator sets and reuses
// every Wout B fragment twice -> halves L2 weight traffic on dominant GEMM.
__global__ __launch_bounds__(256, 1) void k_output(
    const __bf16* __restrict__ h1ctx, const __bf16* __restrict__ h1last,
    const __bf16* __restrict__ imgB, const __bf16* __restrict__ wout,
    const float* __restrict__ bout, const int* __restrict__ answer_lens,
    float* __restrict__ out) {
  __shared__ __bf16 As[32 * KOUT];             // 32 rows x 896 feat (56 KB)
  const int tid = threadIdx.x, wave = tid >> 5, lane = tid & 31;
  const int m0 = blockIdx.x * 32;              // 32 rows share one question n
  const int n  = m0 / TAL;
  const int t0 = m0 % TAL;

  // ---- async-stage A panel into LDS (ASYNCcnt path) ----
#pragma unroll
  for (int it = 0; it < 4; ++it) {             // h1 | ctx : 32 rows x 256
    int idx = tid + it * 256;
    int row = idx >> 5, c = idx & 31;
    async_ld_b128(&As[row * KOUT + c * 8],
                  h1ctx + (size_t)(m0 + row) * 256 + c * 8);
  }
#pragma unroll
  for (int it = 0; it < 8; ++it) {             // img broadcast per row
    int idx = tid + it * 256;
    int row = idx >> 6, c = idx & 63;
    async_ld_b128(&As[row * KOUT + 384 + c * 8], imgB + c * 8);
  }
  if (n > 0) {                                 // history = h1last[n-1]
#pragma unroll
    for (int it = 0; it < 2; ++it) {
      int idx = tid + it * 256;
      int row = idx >> 4, c = idx & 15;
      async_ld_b128(&As[row * KOUT + 256 + c * 8],
                    h1last + (size_t)(n - 1) * HID + c * 8);
    }
  } else {
    for (int i = tid; i < 32 * HID; i += 256) {
      int row = i >> 7, k = i & 127;
      As[row * KOUT + 256 + k] = (__bf16)0.f;
    }
  }
  wait_async0();
  __syncthreads();

  v8f acc0[8], acc1[8];
#pragma unroll
  for (int q = 0; q < 8; ++q) {
    int nt = wave + 8 * q;
    int col = nt * 16 + (lane & 15);
    float bv = (nt < NT_OUT && col < VOC) ? bout[col] : 0.f;
#pragma unroll
    for (int r = 0; r < 8; ++r) { acc0[q][r] = bv; acc1[q][r] = bv; }
  }
  for (int kt = 0; kt < KT_OUT; ++kt) {
    v16bf a0 = load_a_frag(As, KOUT, kt, lane);
    v16bf a1 = load_a_frag(As + 16 * KOUT, KOUT, kt, lane);
    if (kt + 1 < KT_OUT)                       // stream weights through L2
      __builtin_prefetch(wout + (((size_t)((kt + 1) * NT_OUT + wave) * 32 + lane) << 4), 0, 1);
#pragma unroll
    for (int q = 0; q < 8; ++q) {
      int nt = wave + 8 * q;                   // wave-uniform: EXEC stays full
      if (nt < NT_OUT) {
        v16bf b = load_b_frag(wout, kt, nt, NT_OUT, lane);
        acc0[q] = wmma_bf16(a0, b, acc0[q]);
        acc1[q] = wmma_bf16(a1, b, acc1[q]);
      }
    }
  }
  const int alen = answer_lens[n];
#pragma unroll
  for (int q = 0; q < 8; ++q) {
    int nt = wave + 8 * q;
    if (nt >= NT_OUT) continue;
    int col = nt * 16 + (lane & 15);
    if (col >= VOC) continue;
#pragma unroll
    for (int r = 0; r < 8; ++r) {
      int row = r + ((lane >> 4) << 3);
      float v0 = (t0 + row < alen) ? acc0[q][r] : 0.f;
      float v1 = (t0 + 16 + row < alen) ? acc1[q][r] : 0.f;
      out[(size_t)(m0 + row) * VOC + col] = v0;
      out[(size_t)(m0 + 16 + row) * VOC + col] = v1;
    }
  }
}

// ------------------------------- launcher ----------------------------------
extern "C" void kernel_launch(void* const* d_in, const int* in_sizes, int n_in,
                              void* d_out, int out_size, void* d_ws, size_t ws_size,
                              hipStream_t stream) {
  (void)in_sizes; (void)n_in; (void)out_size; (void)ws_size;
  const int*   questions   = (const int*)d_in[0];
  const int*   answers     = (const int*)d_in[1];
  const int*   answer_lens = (const int*)d_in[2];
  const float* img  = (const float*)d_in[3];
  const float* emb  = (const float*)d_in[4];
  const float* eWih0 = (const float*)d_in[5];
  const float* eWhh0 = (const float*)d_in[6];
  const float* eb0   = (const float*)d_in[7];
  const float* eWih1 = (const float*)d_in[8];
  const float* eWhh1 = (const float*)d_in[9];
  const float* eb1   = (const float*)d_in[10];
  const float* dWih0 = (const float*)d_in[11];
  const float* dWhh0 = (const float*)d_in[12];
  const float* db0   = (const float*)d_in[13];
  const float* dWih1 = (const float*)d_in[14];
  const float* dWhh1 = (const float*)d_in[15];
  const float* db1   = (const float*)d_in[16];
  const float* Wk    = (const float*)d_in[17];
  const float* bk    = (const float*)d_in[18];
  const float* Wout  = (const float*)d_in[19];
  const float* bout  = (const float*)d_in[20];

  char* ws = (char*)d_ws;
  size_t off = 0;
  auto alloc = [&](size_t bytes) -> char* {
    char* p = ws + off;
    off = (off + bytes + 255) & ~(size_t)255;
    return p;
  };
  __bf16* embB  = (__bf16*)alloc((size_t)VOC * HID * 2);
  __bf16* imgB  = (__bf16*)alloc((size_t)FDIM * 2);
  const size_t lstmW = (size_t)4 * 32 * 512 * 2;   // KT=4, NT=32 fragments
  __bf16* sEih0 = (__bf16*)alloc(lstmW);
  __bf16* sEhh0 = (__bf16*)alloc(lstmW);
  __bf16* sEih1 = (__bf16*)alloc(lstmW);
  __bf16* sEhh1 = (__bf16*)alloc(lstmW);
  __bf16* sDih0 = (__bf16*)alloc(lstmW);
  __bf16* sDhh0 = (__bf16*)alloc(lstmW);
  __bf16* sDih1 = (__bf16*)alloc(lstmW);
  __bf16* sDhh1 = (__bf16*)alloc(lstmW);
  __bf16* sWk   = (__bf16*)alloc((size_t)4 * 8 * 512 * 2);
  __bf16* sWout = (__bf16*)alloc((size_t)KT_OUT * NT_OUT * 512 * 2);
  __bf16* outs  = (__bf16*)alloc((size_t)NB * TQL * HID * 2);
  __bf16* keys  = (__bf16*)alloc((size_t)NB * TQL * HID * 2);
  __bf16* ehb0  = (__bf16*)alloc((size_t)NB * HID * 2);
  __bf16* ehb1  = (__bf16*)alloc((size_t)NB * HID * 2);
  float*  ec0   = (float*)alloc((size_t)NB * HID * 4);
  float*  ec1   = (float*)alloc((size_t)NB * HID * 4);
  __bf16* h1ctx = (__bf16*)alloc((size_t)NB * TAL * 256 * 2);
  __bf16* h1last= (__bf16*)alloc((size_t)NB * HID * 2);

  // ---- weight / embedding prep ----
  k_pad<<<(VOC * HID + 255) / 256, 256, 0, stream>>>(emb, embB, VOC, EMB, HID);
  k_pad<<<(FDIM + 255) / 256, 256, 0, stream>>>(img, imgB, 1, FDIM, FDIM);
  k_swizzleW<<<256, 256, 0, stream>>>(eWih0, sEih0, GATES, EMB, 4, 32);
  k_swizzleW<<<256, 256, 0, stream>>>(eWhh0, sEhh0, GATES, HID, 4, 32);
  k_swizzleW<<<256, 256, 0, stream>>>(eWih1, sEih1, GATES, HID, 4, 32);
  k_swizzleW<<<256, 256, 0, stream>>>(eWhh1, sEhh1, GATES, HID, 4, 32);
  k_swizzleW<<<256, 256, 0, stream>>>(dWih0, sDih0, GATES, EMB, 4, 32);
  k_swizzleW<<<256, 256, 0, stream>>>(dWhh0, sDhh0, GATES, HID, 4, 32);
  k_swizzleW<<<256, 256, 0, stream>>>(dWih1, sDih1, GATES, HID, 4, 32);
  k_swizzleW<<<256, 256, 0, stream>>>(dWhh1, sDhh1, GATES, HID, 4, 32);
  k_swizzleW<<<64, 256, 0, stream>>>(Wk, sWk, HID, HID, 4, 8);
  k_swizzleW<<<(KT_OUT * NT_OUT * 512 + 255) / 256, 256, 0, stream>>>(
      Wout, sWout, VOC, KOUT, KT_OUT, NT_OUT);

  // ---- model ----
  k_encoder<<<NB / 16, 256, 0, stream>>>(questions, embB,
      sEih0, sEhh0, eb0, sEih1, sEhh1, eb1, outs, ehb0, ec0, ehb1, ec1);
  k_keys<<<NB * TQL / (16 * 8), 256, 0, stream>>>(outs, sWk, bk, keys);
  k_decoder<<<NB / 16, 256, 0, stream>>>(answers, embB,
      sDih0, sDhh0, db0, sDih1, sDhh1, db1,
      ehb0, ec0, ehb1, ec1, keys, h1ctx, h1last);
  k_output<<<NB * TAL / 32, 256, 0, stream>>>(h1ctx, h1last, imgB, sWout, bout,
                                              answer_lens, (float*)d_out);
}
